// TransformerBlock_55705725829400
// MI455X (gfx1250) — matrix operand order repaired and checked by reading, not compiled, not run on hardware
//
#include <hip/hip_runtime.h>
#include <hip/hip_bf16.h>
#include <stdint.h>

#define T_SEQ 4096
#define C_DIM 1024
#define HF_DIM 4096
#define NH 16

typedef __attribute__((ext_vector_type(16))) __bf16 v16bf;
typedef __attribute__((ext_vector_type(8)))  float  v8f;

union AFrag { unsigned int u[8]; v16bf v; };
union BFrag { unsigned int u[8]; v16bf v; };
union U128  { uint4 q; __bf16 e[8]; };

__device__ __forceinline__ v8f wmma_bf16(const AFrag& a, const BFrag& b, v8f c) {
  return __builtin_amdgcn_wmma_f32_16x16x32_bf16(false, a.v, false, b.v,
                                                 (short)0, c, false, false);
}

// ---- CDNA5 async global->LDS copy (ASYNCcnt-tracked DMA, no VGPR round-trip) ----
__device__ __forceinline__ void async_b128(const void* lds_p, const void* g_p) {
  asm volatile("global_load_async_to_lds_b128 %0, %1, off"
               :: "v"((unsigned)(uintptr_t)lds_p),
                  "v"((unsigned long long)(uintptr_t)g_p)
               : "memory");
}
template <int N>
__device__ __forceinline__ void s_wait_async() {
  asm volatile("s_wait_asynccnt %0" :: "n"(N) : "memory");
}

// A fragment (16x32 bf16, MxK): lane m = lane&15; VGPR v: k = kb+2v,{+1};
// kb = 16 for lanes 16-31. Pairs are contiguous dwords.
__device__ __forceinline__ void load_a(AFrag& f, const __bf16* base, int stride) {
  const int lane = threadIdx.x & 31;
  const unsigned int* p = (const unsigned int*)(base + (size_t)(lane & 15) * stride);
  const int o = (lane >> 4) ? 8 : 0;
#pragma unroll
  for (int i = 0; i < 8; ++i) f.u[i] = p[o + i];
}

// B fragment (32x16 bf16, KxN) from memory holding B^T rows: n = lane&15,
// kb = 16 for lanes 16-31; VGPR v: k = kb+2v,{+1}.  base[n*stride + k].
__device__ __forceinline__ void load_b(BFrag& f, const __bf16* base, int stride) {
  const int lane = threadIdx.x & 31;
  const unsigned int* p = (const unsigned int*)(base + (size_t)(lane & 15) * stride);
  const int o = (lane >> 4) ? 8 : 0;
#pragma unroll
  for (int i = 0; i < 8; ++i) f.u[i] = p[o + i];
}

__global__ void cvt_bf16_kernel(const float* __restrict__ s, __bf16* __restrict__ d, int n) {
  int i = blockIdx.x * 256 + threadIdx.x;
  int stride = gridDim.x * 256;
  for (; i < n; i += stride) d[i] = (__bf16)s[i];
}

__global__ __launch_bounds__(256) void rmsnorm_kernel(const float* __restrict__ x,
                                                      const float* __restrict__ w,
                                                      __bf16* __restrict__ out) {
  __shared__ float red[256];
  const int row = blockIdx.x, tid = threadIdx.x;
  const float* xr = x + (size_t)row * C_DIM;
  float v[4]; float ss = 0.f;
#pragma unroll
  for (int i = 0; i < 4; ++i) { v[i] = xr[tid + i * 256]; ss += v[i] * v[i]; }
  red[tid] = ss; __syncthreads();
  for (int s = 128; s > 0; s >>= 1) { if (tid < s) red[tid] += red[tid + s]; __syncthreads(); }
  const float inv = rsqrtf(red[0] * (1.0f / C_DIM) + 1e-6f);
#pragma unroll
  for (int i = 0; i < 4; ++i) {
    const int c = tid + i * 256;
    out[(size_t)row * C_DIM + c] = (__bf16)(v[i] * inv * w[c]);
  }
}

// C[M,N] = A[M,K](bf16,rm) x W[N,K](bf16,rm)^T ; MODE 0: bf16 out, MODE 1: f32 out + res
// Double-buffered LDS tiles filled by async global->LDS DMA; last step peeled so the
// steady-state loop is branch-free with a constant s_wait_asynccnt threshold.
template <int MODE>
__global__ __launch_bounds__(256) void gemm_bf16_kernel(
    const __bf16* __restrict__ A, const __bf16* __restrict__ W,
    void* __restrict__ out, const float* __restrict__ res, int M, int N, int K) {
  __shared__ __bf16 At[2][128 * 40];
  __shared__ __bf16 Bt[2][128 * 40];
  const int tid = threadIdx.x, lane = tid & 31, w = tid >> 5;
  const int wm = w >> 1, wn = w & 1;
  const int Mb = blockIdx.y * 128, Nb = blockIdx.x * 128;
  v8f acc[2][4] = {};
  const int r = tid >> 1, cb = (tid & 1) * 16;

  const __bf16* gA = A + (size_t)(Mb + r) * K + cb;
  const __bf16* gW = W + (size_t)(Nb + r) * K + cb;
  __bf16* sA = &At[0][r * 40 + cb];
  __bf16* sB = &Bt[0][r * 40 + cb];
  const size_t bufstep = 128 * 40;

  auto issue = [&](int buf, int k0) {
    async_b128(sA + buf * bufstep,     gA + k0);
    async_b128(sA + buf * bufstep + 8, gA + k0 + 8);
    async_b128(sB + buf * bufstep,     gW + k0);
    async_b128(sB + buf * bufstep + 8, gW + k0 + 8);
  };
  auto compute = [&](int buf) {
    AFrag a[2]; BFrag b[4];
#pragma unroll
    for (int mt = 0; mt < 2; ++mt) load_a(a[mt], &At[buf][(wm * 32 + mt * 16) * 40], 40);
#pragma unroll
    for (int nt = 0; nt < 4; ++nt) load_b(b[nt], &Bt[buf][(wn * 64 + nt * 16) * 40], 40);
#pragma unroll
    for (int nt = 0; nt < 4; ++nt)
#pragma unroll
      for (int mt = 0; mt < 2; ++mt) acc[mt][nt] = wmma_bf16(a[mt], b[nt], acc[mt][nt]);
  };

  issue(0, 0);
  int cur = 0;
  for (int k0 = 0; k0 < K - 32; k0 += 32) {   // steady state: branch-free
    issue(cur ^ 1, k0 + 32);
    s_wait_async<4>();
    __syncthreads();
    compute(cur);
    __syncthreads();
    cur ^= 1;
  }
  s_wait_async<0>();                          // peeled last step
  __syncthreads();
  compute(cur);

  const int mhi = (lane >> 4) * 8, nc = lane & 15;
#pragma unroll
  for (int mt = 0; mt < 2; ++mt)
#pragma unroll
    for (int nt = 0; nt < 4; ++nt)
#pragma unroll
      for (int v = 0; v < 8; ++v) {
        const size_t row = Mb + wm * 32 + mt * 16 + mhi + v;
        const size_t col = Nb + wn * 64 + nt * 16 + nc;
        const float val = acc[mt][nt][v];
        if (MODE == 0) ((__bf16*)out)[row * N + col] = (__bf16)val;
        else           ((float*)out)[row * N + col] = val + res[row * N + col];
      }
}

// Fused U = silu(A W1^T) * (A W3^T), bf16 out; double-buffered async staging.
__global__ __launch_bounds__(256) void ffn13_kernel(
    const __bf16* __restrict__ A, const __bf16* __restrict__ W1,
    const __bf16* __restrict__ W3, __bf16* __restrict__ U, int M, int N, int K) {
  __shared__ __bf16 At[2][128 * 40];
  __shared__ __bf16 B1[2][128 * 40];
  __shared__ __bf16 B3[2][128 * 40];
  const int tid = threadIdx.x, lane = tid & 31, w = tid >> 5;
  const int wm = w >> 1, wn = w & 1;
  const int Mb = blockIdx.y * 128, Nb = blockIdx.x * 128;
  v8f acc1[2][4] = {};
  v8f acc3[2][4] = {};
  const int r = tid >> 1, cb = (tid & 1) * 16;

  const __bf16* gA = A  + (size_t)(Mb + r) * K + cb;
  const __bf16* g1 = W1 + (size_t)(Nb + r) * K + cb;
  const __bf16* g3 = W3 + (size_t)(Nb + r) * K + cb;
  __bf16* sA = &At[0][r * 40 + cb];
  __bf16* s1 = &B1[0][r * 40 + cb];
  __bf16* s3 = &B3[0][r * 40 + cb];
  const size_t bufstep = 128 * 40;

  auto issue = [&](int buf, int k0) {
    async_b128(sA + buf * bufstep,     gA + k0);
    async_b128(sA + buf * bufstep + 8, gA + k0 + 8);
    async_b128(s1 + buf * bufstep,     g1 + k0);
    async_b128(s1 + buf * bufstep + 8, g1 + k0 + 8);
    async_b128(s3 + buf * bufstep,     g3 + k0);
    async_b128(s3 + buf * bufstep + 8, g3 + k0 + 8);
  };
  auto compute = [&](int buf) {
    AFrag a[2]; BFrag b1[4], b3[4];
#pragma unroll
    for (int mt = 0; mt < 2; ++mt) load_a(a[mt], &At[buf][(wm * 32 + mt * 16) * 40], 40);
#pragma unroll
    for (int nt = 0; nt < 4; ++nt) {
      load_b(b1[nt], &B1[buf][(wn * 64 + nt * 16) * 40], 40);
      load_b(b3[nt], &B3[buf][(wn * 64 + nt * 16) * 40], 40);
    }
#pragma unroll
    for (int nt = 0; nt < 4; ++nt)
#pragma unroll
      for (int mt = 0; mt < 2; ++mt) {
        acc1[mt][nt] = wmma_bf16(a[mt], b1[nt], acc1[mt][nt]);
        acc3[mt][nt] = wmma_bf16(a[mt], b3[nt], acc3[mt][nt]);
      }
  };

  issue(0, 0);
  int cur = 0;
  for (int k0 = 0; k0 < K - 32; k0 += 32) {
    issue(cur ^ 1, k0 + 32);
    s_wait_async<6>();
    __syncthreads();
    compute(cur);
    __syncthreads();
    cur ^= 1;
  }
  s_wait_async<0>();
  __syncthreads();
  compute(cur);

  const int mhi = (lane >> 4) * 8, nc = lane & 15;
#pragma unroll
  for (int mt = 0; mt < 2; ++mt)
#pragma unroll
    for (int nt = 0; nt < 4; ++nt)
#pragma unroll
      for (int v = 0; v < 8; ++v) {
        const size_t row = Mb + wm * 32 + mt * 16 + mhi + v;
        const size_t col = Nb + wn * 64 + nt * 16 + nc;
        const float a1 = acc1[mt][nt][v];
        const float a3 = acc3[mt][nt][v];
        const float sv = a1 / (1.0f + __expf(-a1));
        U[row * N + col] = (__bf16)(sv * a3);
      }
}

// Flash attention: qkv [T, 3C] bf16; each wave owns a 16-row Q tile of one head;
// 8 waves/block share one head's K/V tiles (128 query rows per block).
__global__ __launch_bounds__(256) void attn_kernel(const __bf16* __restrict__ qkv,
                                                   __bf16* __restrict__ y) {
  __shared__ __bf16 Kt[32 * 72];       // [key][c]
  __shared__ __bf16 Vt[64 * 40];       // transposed: [c][key]
  __shared__ __bf16 Pb[8][16 * 40];    // per-wave P scratch (C->A layout reshape)
  const int tid = threadIdx.x, lane = tid & 31, w = tid >> 5;
  const int h  = blockIdx.x >> 5;          // 32 q-blocks per head
  const int qb = (blockIdx.x & 31) * 128;
  const int qw = qb + w * 16;
  const int mhi = (lane >> 4) * 8, nc = lane & 15;
  const float NEG_INF = -__builtin_inff();

  AFrag q0, q1;
  load_a(q0, qkv + (size_t)qw * 3072 + h * 64,      3072);
  load_a(q1, qkv + (size_t)qw * 3072 + h * 64 + 32, 3072);

  float mrow[8], lrow[8];
#pragma unroll
  for (int v = 0; v < 8; ++v) { mrow[v] = NEG_INF; lrow[v] = 0.f; }
  v8f acc[4] = {};

  const int nkt = (qb + 128) >> 5;
  const int lr = tid >> 3, lc = (tid & 7) * 8;
  for (int kt = 0; kt < nkt; ++kt) {
    const int kb = kt * 32;
    // stage K tile [32x64] via async DMA (LDS byte addr 144*lr + 16*(tid&7): 16B aligned)
    async_b128(Kt + lr * 72 + lc,
               qkv + (size_t)(kb + lr) * 3072 + 1024 + h * 64 + lc);
    // stage V tile transposed [64x32] (scatter, must go through VGPRs)
    U128 uv;
    uv.q = *(const uint4*)(qkv + (size_t)(kb + lr) * 3072 + 2048 + h * 64 + lc);
#pragma unroll
    for (int i = 0; i < 8; ++i) Vt[(lc + i) * 40 + lr] = uv.e[i];
    s_wait_async<0>();
    __syncthreads();

    if (kb <= qw + 15) {            // causal: tile has live keys for this wave
      const int o = (lane >> 4) ? 8 : 0;
      // batch all K fragments, then run the S-WMMA chain
      BFrag bk[2][2];
#pragma unroll
      for (int nt = 0; nt < 2; ++nt) {
        const unsigned int* pk = (const unsigned int*)(Kt + (nt * 16 + nc) * 72);
#pragma unroll
        for (int i = 0; i < 8; ++i) { bk[nt][0].u[i] = pk[o + i]; bk[nt][1].u[i] = pk[16 + o + i]; }
      }
      v8f s[2];
#pragma unroll
      for (int nt = 0; nt < 2; ++nt) {
        v8f c = {};
        c = wmma_bf16(q0, bk[nt][0], c);
        c = wmma_bf16(q1, bk[nt][1], c);
        s[nt] = c;
      }
      // online softmax; row m = mhi+v lives across lanes sharing (lane&16)
#pragma unroll
      for (int v = 0; v < 8; ++v) {
        const int qrow = qw + mhi + v;
        float s0 = s[0][v] * 0.125f;          // 1/sqrt(64)
        float s1 = s[1][v] * 0.125f;
        if (kb + nc > qrow)      s0 = NEG_INF;
        if (kb + 16 + nc > qrow) s1 = NEG_INF;
        float mx = fmaxf(s0, s1);
#pragma unroll
        for (int d = 1; d < 16; d <<= 1) mx = fmaxf(mx, __shfl_xor(mx, d, 16));
        const float mn = fmaxf(mrow[v], mx);
        const float sc = __expf(mrow[v] - mn);
        const float e0 = __expf(s0 - mn);
        const float e1 = __expf(s1 - mn);
        float rs = e0 + e1;
#pragma unroll
        for (int d = 1; d < 16; d <<= 1) rs += __shfl_xor(rs, d, 16);
        lrow[v] = lrow[v] * sc + rs;
        mrow[v] = mn;
#pragma unroll
        for (int t = 0; t < 4; ++t) acc[t][v] *= sc;
        Pb[w][(mhi + v) * 40 + nc]      = (__bf16)e0;
        Pb[w][(mhi + v) * 40 + 16 + nc] = (__bf16)e1;
      }
      AFrag pf;
      load_a(pf, &Pb[w][0], 40);
      BFrag bv[4];
#pragma unroll
      for (int nt = 0; nt < 4; ++nt) {
        const unsigned int* pv = (const unsigned int*)(Vt + (nt * 16 + nc) * 40);
#pragma unroll
        for (int i = 0; i < 8; ++i) bv[nt].u[i] = pv[o + i];
      }
#pragma unroll
      for (int nt = 0; nt < 4; ++nt) acc[nt] = wmma_bf16(pf, bv[nt], acc[nt]);
    }
    __syncthreads();
  }
#pragma unroll
  for (int nt = 0; nt < 4; ++nt)
#pragma unroll
    for (int v = 0; v < 8; ++v) {
      const size_t row = qw + mhi + v;
      const size_t col = h * 64 + nt * 16 + nc;
      y[row * 1024 + col] = (__bf16)(acc[nt][v] / lrow[v]);
    }
}

extern "C" void kernel_launch(void* const* d_in, const int* in_sizes, int n_in,
                              void* d_out, int out_size, void* d_ws, size_t ws_size,
                              hipStream_t stream) {
  (void)in_sizes; (void)n_in; (void)out_size; (void)ws_size;
  const float* x           = (const float*)d_in[0];
  const float* attn_norm_w = (const float*)d_in[1];
  const float* ffn_norm_w  = (const float*)d_in[2];
  const float* c_attn_w    = (const float*)d_in[3];
  const float* c_proj_w    = (const float*)d_in[4];
  const float* w1          = (const float*)d_in[5];
  const float* w2          = (const float*)d_in[6];
  const float* w3          = (const float*)d_in[7];
  float* out = (float*)d_out;

  const size_t T = T_SEQ, C = C_DIM, HF = HF_DIM;
  char* ws = (char*)d_ws;
  size_t o = 0;
  auto take = [&](size_t bytes) -> char* {
    char* p = ws + o;
    o += (bytes + 255) & ~(size_t)255;
    return p;
  };
  __bf16* wattn = (__bf16*)take(3 * C * C * 2);
  __bf16* wproj = (__bf16*)take(C * C * 2);
  __bf16* w1b   = (__bf16*)take(HF * C * 2);
  __bf16* w3b   = (__bf16*)take(HF * C * 2);
  __bf16* w2b   = (__bf16*)take(C * HF * 2);
  __bf16* xn    = (__bf16*)take(T * C * 2);
  __bf16* qkvu  = (__bf16*)take(T * HF * 2);   // holds qkv [T,3C] then u [T,HF]
  __bf16* yat   = (__bf16*)take(T * C * 2);
  float*  hbuf  = (float*) take(T * C * 4);

  const dim3 blk(256);
  // weights -> bf16
  cvt_bf16_kernel<<<dim3(4096), blk, 0, stream>>>(c_attn_w, wattn, (int)(3 * C * C));
  cvt_bf16_kernel<<<dim3(4096), blk, 0, stream>>>(c_proj_w, wproj, (int)(C * C));
  cvt_bf16_kernel<<<dim3(4096), blk, 0, stream>>>(w1, w1b, (int)(HF * C));
  cvt_bf16_kernel<<<dim3(4096), blk, 0, stream>>>(w3, w3b, (int)(HF * C));
  cvt_bf16_kernel<<<dim3(4096), blk, 0, stream>>>(w2, w2b, (int)(C * HF));
  // attn path
  rmsnorm_kernel<<<dim3(T_SEQ), blk, 0, stream>>>(x, attn_norm_w, xn);
  gemm_bf16_kernel<0><<<dim3(3 * C_DIM / 128, T_SEQ / 128), blk, 0, stream>>>(
      xn, wattn, qkvu, nullptr, T_SEQ, 3 * C_DIM, C_DIM);
  attn_kernel<<<dim3(NH * (T_SEQ / 128)), blk, 0, stream>>>(qkvu, yat);
  gemm_bf16_kernel<1><<<dim3(C_DIM / 128, T_SEQ / 128), blk, 0, stream>>>(
      yat, wproj, hbuf, x, T_SEQ, C_DIM, C_DIM);
  // ffn path
  rmsnorm_kernel<<<dim3(T_SEQ), blk, 0, stream>>>(hbuf, ffn_norm_w, xn);
  ffn13_kernel<<<dim3(HF_DIM / 128, T_SEQ / 128), blk, 0, stream>>>(
      xn, w1b, w3b, qkvu, T_SEQ, HF_DIM, C_DIM);
  gemm_bf16_kernel<1><<<dim3(C_DIM / 128, T_SEQ / 128), blk, 0, stream>>>(
      qkvu, w2b, out, hbuf, T_SEQ, C_DIM, HF_DIM);
}
